// CirculantFFTDense_9440338117194
// MI455X (gfx1250) — matrix operand order repaired and checked by reading, not compiled, not run on hardware
//
#include <hip/hip_runtime.h>

// Circulant matmul out[m,n] = sum_k x[m,k] * w[(n-k) mod N] as a WMMA bf16x3
// (error-compensated) GEMM on gfx1250, with async global->LDS staging and a
// rotating circulant-B buffer.

typedef __attribute__((ext_vector_type(16))) __bf16 bf16x16;
typedef __attribute__((ext_vector_type(8)))  float  f32x8;

#define N_DIM 4096
#define BATCH 16384
#define BM 128
#define BN 128
#define BK 32
#define LDT 32            // LDS row stride in ushorts (64 B); banks handled by XOR swizzle

__device__ __forceinline__ unsigned short f32_to_bf16(float f) {
  unsigned int u = __float_as_uint(f);
  u += 0x7fffu + ((u >> 16) & 1u);          // round-to-nearest-even
  return (unsigned short)(u >> 16);
}
__device__ __forceinline__ float bf16_to_f32(unsigned short h) {
  return __uint_as_float(((unsigned int)h) << 16);
}
__device__ __forceinline__ unsigned lds_off32(const void* p) {
  // generic pointer to LDS: low 32 bits are the LDS byte offset
  return (unsigned)(unsigned long long)(uintptr_t)p;
}
// GLOBAL_LOAD_ASYNC_TO_LDS_B128: VDST = per-lane LDS byte offset, VADDR = 64b
// global address (GV mode). Tracked by ASYNCcnt.
__device__ __forceinline__ void async_copy_b128(unsigned lds_byte, const void* gptr) {
  unsigned long long ga = (unsigned long long)(uintptr_t)gptr;
  asm volatile("global_load_async_to_lds_b128 %0, %1, off"
               :: "v"(lds_byte), "v"(ga)
               : "memory");
}

// wr2[i] = w[(-i) mod N], i in [0, 8192): reversed & period-doubled bf16 hi/lo
// tables so B^T[n][k] = w[(n-k) mod N] = wr2[k - n + N] is contiguous in k.
__global__ void circ_prep_kernel(const float* __restrict__ w,
                                 unsigned short* __restrict__ wr2h,
                                 unsigned short* __restrict__ wr2l) {
  int i = blockIdx.x * blockDim.x + threadIdx.x;        // 0..8191
  int src = (N_DIM - (i & (N_DIM - 1))) & (N_DIM - 1);
  float f = w[src];
  unsigned short h = f32_to_bf16(f);
  unsigned short l = f32_to_bf16(f - bf16_to_f32(h));
  wr2h[i] = h;
  wr2l[i] = l;
}

__global__ __launch_bounds__(256)
void circ_gemm_kernel(const float* __restrict__ x,
                      const unsigned short* __restrict__ wr2h,
                      const unsigned short* __restrict__ wr2l,
                      float* __restrict__ out) {
  __shared__ __align__(16) float          Xs[2][BM * BK];   // raw fp32 stage, 2 x 16 KB
  __shared__ __align__(16) unsigned short Ah[BM * LDT];     // 8 KB each
  __shared__ __align__(16) unsigned short Al[BM * LDT];
  __shared__ __align__(16) unsigned short Bh[BN * LDT];     // physical rows p=(n-k0)&127
  __shared__ __align__(16) unsigned short Bl[BN * LDT];

  const int tid  = threadIdx.x;
  const int lane = tid & 31;
  const int wave = tid >> 5;     // 0..7
  const int wm   = wave & 1;     // 2 wave-rows -> 64 M each
  const int wn   = wave >> 1;    // 4 wave-cols -> 32 N each
  const int half = lane >> 4;
  const int lr   = lane & 15;

  const int n0 = blockIdx.x * BN;   // fast grid dim -> blocks sharing x rows adjacent (L2 reuse)
  const int m0 = blockIdx.y * BM;

  f32x8 acc[4][2];
#pragma unroll
  for (int i = 0; i < 4; ++i)
#pragma unroll
    for (int j = 0; j < 2; ++j)
      acc[i][j] = (f32x8)0.0f;

  // ---- prologue: async-copy x tile 0, fully gather circulant B for k0=0 ----
#pragma unroll
  for (int i = 0; i < 4; ++i) {
    int q  = tid + (i << 8);
    int m  = q >> 3;
    int kq = (q & 7) << 2;
    async_copy_b128(lds_off32(&Xs[0][q << 2]),
                    x + (size_t)(m0 + m) * N_DIM + kq);
  }
  {
    int rn  = tid >> 3;            // 0..31
    int rkb = (tid & 7) << 2;      // 0,4,...,28
#pragma unroll
    for (int i = 0; i < 4; ++i) {
      int n    = rn + (i << 5);                       // 0..127, k0 = 0 -> p = n
      int base = N_DIM - (n0 + n) + rkb;              // wr2 index at k=rkb
      unsigned short h0 = wr2h[base + 0], h1 = wr2h[base + 1];
      unsigned short h2 = wr2h[base + 2], h3 = wr2h[base + 3];
      unsigned short l0 = wr2l[base + 0], l1 = wr2l[base + 1];
      unsigned short l2 = wr2l[base + 2], l3 = wr2l[base + 3];
      int col = ((((rkb >> 3) ^ (n & 3)) << 3) | (rkb & 7));
      uint2 ph, pl;
      ph.x = (unsigned)h0 | ((unsigned)h1 << 16);
      ph.y = (unsigned)h2 | ((unsigned)h3 << 16);
      pl.x = (unsigned)l0 | ((unsigned)l1 << 16);
      pl.y = (unsigned)l2 | ((unsigned)l3 << 16);
      *(uint2*)&Bh[n * LDT + col] = ph;
      *(uint2*)&Bl[n * LDT + col] = pl;
    }
  }

  for (int k0 = 0; k0 < N_DIM; k0 += BK) {
    const int buf = (k0 >> 5) & 1;

    // ---- issue async copy of NEXT x tile into the other buffer ----
    if (k0 + BK < N_DIM) {
#pragma unroll
      for (int i = 0; i < 4; ++i) {
        int q  = tid + (i << 8);
        int m  = q >> 3;
        int kq = (q & 7) << 2;
        async_copy_b128(lds_off32(&Xs[buf ^ 1][q << 2]),
                        x + (size_t)(m0 + m) * N_DIM + (k0 + BK) + kq);
      }
    }

    // ---- refresh only the 32 stale circulant rows (n in [0,32)) ----
    if (k0 > 0) {
      int n    = tid >> 3;                            // logical row 0..31
      int rkb  = (tid & 7) << 2;
      int p    = (n - k0) & 127;                      // physical row
      int base = k0 - (n0 + n) + N_DIM + rkb;
      unsigned short h0 = wr2h[base + 0], h1 = wr2h[base + 1];
      unsigned short h2 = wr2h[base + 2], h3 = wr2h[base + 3];
      unsigned short l0 = wr2l[base + 0], l1 = wr2l[base + 1];
      unsigned short l2 = wr2l[base + 2], l3 = wr2l[base + 3];
      int col = ((((rkb >> 3) ^ (p & 3)) << 3) | (rkb & 7));
      uint2 ph, pl;
      ph.x = (unsigned)h0 | ((unsigned)h1 << 16);
      ph.y = (unsigned)h2 | ((unsigned)h3 << 16);
      pl.x = (unsigned)l0 | ((unsigned)l1 << 16);
      pl.y = (unsigned)l2 | ((unsigned)l3 << 16);
      *(uint2*)&Bh[p * LDT + col] = ph;
      *(uint2*)&Bl[p * LDT + col] = pl;
    }

    // ---- wait for CURRENT tile's async copies (next tile stays in flight) ----
    if (k0 + BK < N_DIM) asm volatile("s_wait_asynccnt 0x4" ::: "memory");
    else                 asm volatile("s_wait_asynccnt 0x0" ::: "memory");

    // ---- convert own fp32 chunks -> bf16 hi/lo in Ah/Al (XOR-swizzled) ----
#pragma unroll
    for (int i = 0; i < 4; ++i) {
      int q  = tid + (i << 8);
      int m  = q >> 3;
      int kq = (q & 7) << 2;
      const float4 v = *(const float4*)&Xs[buf][q << 2];
      unsigned short h0 = f32_to_bf16(v.x), h1 = f32_to_bf16(v.y);
      unsigned short h2 = f32_to_bf16(v.z), h3 = f32_to_bf16(v.w);
      unsigned short l0 = f32_to_bf16(v.x - bf16_to_f32(h0));
      unsigned short l1 = f32_to_bf16(v.y - bf16_to_f32(h1));
      unsigned short l2 = f32_to_bf16(v.z - bf16_to_f32(h2));
      unsigned short l3 = f32_to_bf16(v.w - bf16_to_f32(h3));
      int col = ((((kq >> 3) ^ (m & 3)) << 3) | (kq & 7));
      uint2 ph, pl;
      ph.x = (unsigned)h0 | ((unsigned)h1 << 16);
      ph.y = (unsigned)h2 | ((unsigned)h3 << 16);
      pl.x = (unsigned)l0 | ((unsigned)l1 << 16);
      pl.y = (unsigned)l2 | ((unsigned)l3 << 16);
      *(uint2*)&Ah[m * LDT + col] = ph;
      *(uint2*)&Al[m * LDT + col] = pl;
    }
    __syncthreads();

    // ---- compute: 4x2 tiles, 3 WMMAs each (bf16x3 compensation) ----
    bf16x16 bhf[2], blf[2];
#pragma unroll
    for (int tn = 0; tn < 2; ++tn) {
      int bcol = wn * 32 + tn * 16 + lr;
      int p    = (bcol - k0) & 127;
      int sw   = p & 3;
      int c0   = (half ^ sw) << 3;               // chunk half
      int c1   = ((half + 2) ^ sw) << 3;         // chunk half+2
      union { uint4 q[2]; bf16x16 v; } ub, vb;
      ub.q[0] = *(const uint4*)&Bh[p * LDT + c0];
      ub.q[1] = *(const uint4*)&Bh[p * LDT + c1];
      vb.q[0] = *(const uint4*)&Bl[p * LDT + c0];
      vb.q[1] = *(const uint4*)&Bl[p * LDT + c1];
      bhf[tn] = ub.v;
      blf[tn] = vb.v;
    }
#pragma unroll
    for (int tm = 0; tm < 4; ++tm) {
      int arow = wm * 64 + tm * 16 + lr;
      int sw   = arow & 3;
      int c0   = (half ^ sw) << 3;               // chunk half      (K 0..7 / 8..15)
      int c1   = ((half + 2) ^ sw) << 3;         // chunk half+2    (K 16..23 / 24..31)
      union { uint4 q[2]; bf16x16 v; } ua, va;
      ua.q[0] = *(const uint4*)&Ah[arow * LDT + c0];
      ua.q[1] = *(const uint4*)&Ah[arow * LDT + c1];
      va.q[0] = *(const uint4*)&Al[arow * LDT + c0];
      va.q[1] = *(const uint4*)&Al[arow * LDT + c1];
      bf16x16 ah = ua.v, al = va.v;
#pragma unroll
      for (int tn = 0; tn < 2; ++tn) {
        acc[tm][tn] = __builtin_amdgcn_wmma_f32_16x16x32_bf16(
            false, ah, false, bhf[tn], (short)0, acc[tm][tn], false, false);
        acc[tm][tn] = __builtin_amdgcn_wmma_f32_16x16x32_bf16(
            false, ah, false, blf[tn], (short)0, acc[tm][tn], false, false);
        acc[tm][tn] = __builtin_amdgcn_wmma_f32_16x16x32_bf16(
            false, al, false, bhf[tn], (short)0, acc[tm][tn], false, false);
      }
    }
    __syncthreads();
  }

  // ---- store: 16x16 f32 C layout (lanes<16 rows 0..7, lanes>=16 rows 8..15) ----
#pragma unroll
  for (int tm = 0; tm < 4; ++tm) {
    int mb = m0 + wm * 64 + tm * 16 + half * 8;
#pragma unroll
    for (int tn = 0; tn < 2; ++tn) {
      int nc = n0 + wn * 32 + tn * 16 + lr;
#pragma unroll
      for (int r = 0; r < 8; ++r)
        out[(size_t)(mb + r) * N_DIM + nc] = acc[tm][tn][r];
    }
  }
}

extern "C" void kernel_launch(void* const* d_in, const int* in_sizes, int n_in,
                              void* d_out, int out_size, void* d_ws, size_t ws_size,
                              hipStream_t stream) {
  const float* x = (const float*)d_in[0];        // (16384, 4096) fp32
  const float* w = (const float*)d_in[1];        // (4096,) fp32
  float* out = (float*)d_out;                    // (16384, 4096) fp32

  unsigned short* wr2h = (unsigned short*)d_ws;  // 8192 ushorts
  unsigned short* wr2l = wr2h + 8192;            // 8192 ushorts (32 KB total)

  circ_prep_kernel<<<32, 256, 0, stream>>>(w, wr2h, wr2l);

  dim3 grid(N_DIM / BN, BATCH / BM);             // (32, 128)
  circ_gemm_kernel<<<grid, 256, 0, stream>>>(x, wr2h, wr2l, out);
}
